// ScaledDotProductAttention_455266533556
// MI455X (gfx1250) — matrix operand order repaired and checked by reading, not compiled, not run on hardware
//
#include <hip/hip_runtime.h>

typedef __attribute__((ext_vector_type(16))) _Float16 v16h;
typedef __attribute__((ext_vector_type(8)))  float    v8f;
typedef __attribute__((ext_vector_type(4)))  unsigned int v4u;
typedef __attribute__((ext_vector_type(8)))  int      v8i;
typedef __attribute__((ext_vector_type(4)))  int      v4i;

#define L_SEQ 512
#define D_HEAD 64
#define BH_TOT 128
#define TEMP_INV 0.125f
#define MASK_SENT (-10000.0f)
#define BIG_NEG 1.0e9f

#define KV_PITCH 72          // halves per K/V row (bank-conflict-free: 36 mod 64)
#define P_PITCH  520         // halves per P row  (260 mod 64 = 4)

// LDS layout (bytes)
#define OFF_K  0                         // _Float16 [512][72] = 73728
#define OFF_V  73728                     // _Float16 [512][72] = 73728
#define OFF_Q  147456                    // _Float16 [16][72]  = 2304
#define OFF_M  149760                    // float    [512]     = 2048
#define OFF_S  151808                    // float    [16][512] = 32768 (contig for TDM)
#define OFF_P  184576                    // _Float16 [16][520] = 16640
#define OFF_O  201216                    // float [2][16][64]  = 8192  ([0] contig for TDM)
#define SMEM_BYTES 209408

// ---- Tensor Data Mover: 2D fp32 tile store, LDS -> global ----------------
// D# group0: count=1 | lds_addr | global_addr(57b) | type=2
// D# group1: data_size=2(4B) @ [17:16]; tensor_dim0 @ [79:48]; tensor_dim1 @
// [111:80]; tile_dim0 @ [127:112]; tile_dim1 @ [143:128]; dim0_stride @ [207:160]
__device__ __forceinline__ void tdm_store_tile_f32(unsigned int lds_byte,
                                                   unsigned long long gaddr,
                                                   int dim0, int rows, int stride0) {
  v4u g0 = { 1u, lds_byte, (unsigned int)gaddr,
             (unsigned int)((gaddr >> 32) & 0x01FFFFFFull) | (2u << 30) };
  v8i g1 = { (int)0x00020000u,
             (int)((unsigned)(dim0 & 0xFFFF) << 16),
             (int)(((unsigned)(dim0 >> 16) & 0xFFFFu) | ((unsigned)(rows & 0xFFFF) << 16)),
             (int)(((unsigned)(rows >> 16) & 0xFFFFu) | ((unsigned)(dim0 & 0xFFFF) << 16)),
             (int)(unsigned)(rows & 0xFFFF),
             stride0, 0, 0 };
  v4i z4 = { 0, 0, 0, 0 };
#if __has_include(<hip/amd_detail/amd_gfx1250_TDM.h>)
  v8i z8 = { 0, 0, 0, 0, 0, 0, 0, 0 };
  __builtin_amdgcn_tensor_store_from_lds(g0, g1, z4, z4, z8, 0);
#else
  __builtin_amdgcn_tensor_store_from_lds(g0, g1, z4, z4, 0);
#endif
}

__global__ __launch_bounds__(256) void sdpa_wmma_kernel(
    const float* __restrict__ q, const float* __restrict__ k,
    const float* __restrict__ v, const float* __restrict__ mask,
    const int* __restrict__ n_head_p,
    float* __restrict__ out, float* __restrict__ attn)
{
  extern __shared__ char smem[];
  _Float16* sK = (_Float16*)(smem + OFF_K);
  _Float16* sV = (_Float16*)(smem + OFF_V);
  _Float16* sQ = (_Float16*)(smem + OFF_Q);
  float*    sM = (float*)   (smem + OFF_M);
  float*    sS = (float*)   (smem + OFF_S);
  _Float16* sP = (_Float16*)(smem + OFF_P);
  float*    sO = (float*)   (smem + OFF_O);

  const int tid  = threadIdx.x;
  const int lane = tid & 31;
  const int wave = tid >> 5;
  const int l15  = lane & 15;          // M in A-frag; N in B/C/D frags
  const int ahi  = lane >> 4;          // half-wave selector
  const int it   = blockIdx.x;
  const int bh   = blockIdx.y;
  const int i0   = it * 16;
  const int nh   = n_head_p[0];
  const int bb   = bh / nh;

  const size_t base = (size_t)bh * L_SEQ * D_HEAD;

  // ---- stage K, V (f32 -> f16, padded rows), Q tile, mask into LDS ----
  {
    const float4* k4 = (const float4*)(k + base);
    const float4* v4 = (const float4*)(v + base);
    for (int idx = tid; idx < L_SEQ * D_HEAD / 4; idx += 256) {
      int j = idx >> 4, dg = idx & 15;
      float4 kv = k4[idx];
      float4 vv = v4[idx];
      union { _Float16 h[4]; uint2 u; } pk, pv;
      pk.h[0] = (_Float16)kv.x; pk.h[1] = (_Float16)kv.y;
      pk.h[2] = (_Float16)kv.z; pk.h[3] = (_Float16)kv.w;
      pv.h[0] = (_Float16)vv.x; pv.h[1] = (_Float16)vv.y;
      pv.h[2] = (_Float16)vv.z; pv.h[3] = (_Float16)vv.w;
      ((uint2*)(sK + j * KV_PITCH))[dg] = pk.u;
      ((uint2*)(sV + j * KV_PITCH))[dg] = pv.u;
    }
    const float4* q4 = (const float4*)(q + base + (size_t)i0 * D_HEAD);
    for (int idx = tid; idx < 16 * D_HEAD / 4; idx += 256) {
      int j = idx >> 4, dg = idx & 15;
      float4 qv = q4[idx];
      union { _Float16 h[4]; uint2 u; } pq;
      pq.h[0] = (_Float16)qv.x; pq.h[1] = (_Float16)qv.y;
      pq.h[2] = (_Float16)qv.z; pq.h[3] = (_Float16)qv.w;
      ((uint2*)(sQ + j * KV_PITCH))[dg] = pq.u;
    }
    for (int j = tid; j < L_SEQ; j += 256) {
      float mv = mask[bb * L_SEQ + j];
      sM[j] = (mv == MASK_SENT) ? BIG_NEG : mv;
    }
  }
  __syncthreads();

  // ---- GEMM1: S[16,512] = Q x K^T * (1/8) - m_row - m_col ----
  {
    // hoist A fragments (Q, invariant over jt) and the row mask
    v16h a0, a1;
#pragma unroll
    for (int h = 0; h < 16; ++h) {
      int ka = (ahi << 4) + h;
      a0[h] = sQ[l15 * KV_PITCH + ka];
      a1[h] = sQ[l15 * KV_PITCH + 32 + ka];
    }
    float rowm[8];
#pragma unroll
    for (int r = 0; r < 8; ++r) rowm[r] = sM[i0 + r + (ahi << 3)];

#pragma unroll
    for (int t = 0; t < 4; ++t) {            // constant trip: no exec masking
      const int jt = wave + 8 * t;
      v16h b0, b1;
      const _Float16* krow = &sK[(jt * 16 + l15) * KV_PITCH + (ahi << 4)];
#pragma unroll
      for (int h = 0; h < 16; ++h) { b0[h] = krow[h]; b1[h] = krow[32 + h]; }
      v8f c = {};
      c = __builtin_amdgcn_wmma_f32_16x16x32_f16(false, a0, false, b0,
                                                 (short)0, c, false, false);
      c = __builtin_amdgcn_wmma_f32_16x16x32_f16(false, a1, false, b1,
                                                 (short)0, c, false, false);
      float mcol = sM[jt * 16 + l15];
#pragma unroll
      for (int r = 0; r < 8; ++r) {
        int mr = r + (ahi << 3);
        sS[mr * L_SEQ + jt * 16 + l15] = c[r] * TEMP_INV - rowm[r] - mcol;
      }
    }
  }
  __syncthreads();

  // ---- softmax (2 rows/wave); P back into sS (f32, contig) + sP (f16) ----
#pragma unroll
  for (int rr = 0; rr < 2; ++rr) {
    const int r = wave * 2 + rr;
    float mx = -1.0e30f;
    for (int j = lane; j < L_SEQ; j += 32) mx = fmaxf(mx, sS[r * L_SEQ + j]);
#pragma unroll
    for (int off = 16; off; off >>= 1) mx = fmaxf(mx, __shfl_xor(mx, off, 32));
    float sum = 0.0f;
    for (int j = lane; j < L_SEQ; j += 32) sum += __expf(sS[r * L_SEQ + j] - mx);
#pragma unroll
    for (int off = 16; off; off >>= 1) sum += __shfl_xor(sum, off, 32);
    float inv = 1.0f / sum;
    for (int j = lane; j < L_SEQ; j += 32) {
      float p = __expf(sS[r * L_SEQ + j] - mx) * inv;
      sS[r * L_SEQ + j] = p;
      sP[r * P_PITCH + j] = (_Float16)p;
    }
  }
  __syncthreads();

  // ---- wave 0: async TDM store of the fp32 attn tile (overlaps GEMM2) ----
  if (__builtin_amdgcn_readfirstlane(wave) == 0) {
    unsigned long long ga =
        (unsigned long long)(uintptr_t)(attn + (size_t)bh * L_SEQ * L_SEQ +
                                        (size_t)i0 * L_SEQ);
    tdm_store_tile_f32((unsigned int)(uintptr_t)(void*)sS, ga,
                       L_SEQ, 16, L_SEQ);
  }

  // ---- GEMM2: O[16,64] = P[16,512] x V[512,64]; 4 N-tiles x 2 K-halves ----
  {
    const int nt = wave & 3;
    const int kh = wave >> 2;
    v8f c = {};
#pragma unroll
    for (int d0i = 0; d0i < 256; d0i += 32) {
      const int d0 = kh * 256 + d0i;
      v16h a, bf;
#pragma unroll
      for (int h = 0; h < 16; ++h) {
        int ka = d0 + (ahi << 4) + h;
        a[h] = sP[l15 * P_PITCH + ka];
      }
#pragma unroll
      for (int h = 0; h < 16; ++h) {
        bf[h] = sV[(d0 + (ahi << 4) + h) * KV_PITCH + nt * 16 + l15];
      }
      c = __builtin_amdgcn_wmma_f32_16x16x32_f16(false, a, false, bf,
                                                 (short)0, c, false, false);
    }
#pragma unroll
    for (int r = 0; r < 8; ++r) {
      int mr = r + (ahi << 3);
      sO[(kh * 16 + mr) * D_HEAD + nt * 16 + l15] = c[r];
    }
  }
  __syncthreads();

  // ---- reduce the two K-half partials in place (sO[0] stays contiguous) ----
  for (int idx = tid; idx < 16 * D_HEAD; idx += 256) {
    sO[idx] = sO[idx] + sO[16 * D_HEAD + idx];
  }
  __syncthreads();

  // ---- wave 0: TDM store of the output tile, then drain TENSORcnt ----
  if (__builtin_amdgcn_readfirstlane(wave) == 0) {
    unsigned long long ga =
        (unsigned long long)(uintptr_t)(out + base + (size_t)i0 * D_HEAD);
    tdm_store_tile_f32((unsigned int)(uintptr_t)(void*)sO, ga,
                       D_HEAD, 16, D_HEAD);
    __builtin_amdgcn_s_wait_tensorcnt(0);
  }
}

extern "C" void kernel_launch(void* const* d_in, const int* in_sizes, int n_in,
                              void* d_out, int out_size, void* d_ws, size_t ws_size,
                              hipStream_t stream) {
  const float* q    = (const float*)d_in[0];
  const float* k    = (const float*)d_in[1];
  const float* v    = (const float*)d_in[2];
  const float* mask = (const float*)d_in[3];
  const int*   nh   = (const int*)d_in[4];

  float* out  = (float*)d_out;                              // [128,512,64]
  float* attn = out + (size_t)BH_TOT * L_SEQ * D_HEAD;      // [128,512,512]

  hipFuncSetAttribute((const void*)sdpa_wmma_kernel,
                      hipFuncAttributeMaxDynamicSharedMemorySize, SMEM_BYTES);

  dim3 grid(L_SEQ / 16, BH_TOT);
  sdpa_wmma_kernel<<<grid, 256, SMEM_BYTES, stream>>>(q, k, v, mask, nh, out, attn);
}
